// SlidingWindowAttention_65910568124844
// MI455X (gfx1250) — matrix operand order repaired and checked by reading, not compiled, not run on hardware
//
#include <hip/hip_runtime.h>

// ---------------------------------------------------------------------------
// Sliding-window attention, bf16 WMMA pipeline for gfx1250 (MI455X).
// B=2, S=4096, E=1024, H=16, W=256, D=64.
// ---------------------------------------------------------------------------

typedef __attribute__((ext_vector_type(16))) __bf16 bf16x16;
typedef __attribute__((ext_vector_type(8)))  float  f32x8;
typedef __attribute__((ext_vector_type(4)))  int    i32x4;

union FragU { bf16x16 v; uint4 q[2]; };
union V8U   { uint4 q; __bf16 b[8]; };
union B4U   { uint2 u; __bf16 b[4]; };

// --------------------------- async global->LDS -----------------------------
#if defined(__gfx1250__) && __has_builtin(__builtin_amdgcn_global_load_async_to_lds_b128)
#define HAS_ASYNC_LDS 1
#else
#define HAS_ASYNC_LDS 0
#endif

// Builtin signature (from hipcc diagnostic): (v4i AS(1)*, v4i AS(3)*, Ii, Ii)
typedef __attribute__((address_space(1))) i32x4 GI4;
typedef __attribute__((address_space(3))) i32x4 LI4;

// 16B global -> LDS copy; async (ASYNCcnt) when the toolchain exposes it.
__device__ __forceinline__ void copy16_g2l(const void* g, void* l) {
#if HAS_ASYNC_LDS
  __builtin_amdgcn_global_load_async_to_lds_b128((GI4*)g, (LI4*)l, 0, 0);
#else
  *(uint4*)l = *(const uint4*)g;
#endif
}

__device__ __forceinline__ void g2l_wait0() {
#if HAS_ASYNC_LDS
#if __has_builtin(__builtin_amdgcn_s_wait_asynccnt)
  __builtin_amdgcn_s_wait_asynccnt(0);
#else
  asm volatile("s_wait_asynccnt 0x0" ::: "memory");
#endif
#endif
}

// --------------------------- WMMA fragment loads ---------------------------
// A-fragment (16x32 bf16, row-major source, row stride ldm elements).
// lane m=lane&15, h=lane>>4; VGPR0-3: k=h*8..h*8+7, VGPR4-7: k=16+h*8..+7.
__device__ __forceinline__ bf16x16 ld_frag_a(const __bf16* base, int ldm, int lane) {
  const int m = lane & 15, h = lane >> 4;
  const __bf16* p = base + m * ldm + h * 8;
  FragU f;
  f.q[0] = *(const uint4*)(p);
  f.q[1] = *(const uint4*)(p + 16);
  return f.v;
}

// B-fragment (32x16 bf16) from a TRANSPOSED store Bt[n][k], row stride ldk.
// lane n=lane&15, h=lane>>4; VGPR0-3: k=h*8..h*8+7, VGPR4-7: k=16+h*8..+7.
__device__ __forceinline__ bf16x16 ld_frag_bt(const __bf16* baseT, int ldk, int lane) {
  const int n = lane & 15, h = lane >> 4;
  const __bf16* p = baseT + n * ldk + h * 8;
  FragU f;
  f.q[0] = *(const uint4*)(p);
  f.q[1] = *(const uint4*)(p + 16);
  return f.v;
}

__device__ __forceinline__ f32x8 wmma_bf16(bf16x16 a, bf16x16 b, f32x8 c) {
  return __builtin_amdgcn_wmma_f32_16x16x32_bf16(false, a, false, b, (short)0, c,
                                                 false, false);
}

// ---------------------------------------------------------------------------
// f32 -> bf16 conversion (vectorized 4-wide)
// ---------------------------------------------------------------------------
__global__ __launch_bounds__(256)
void cvt_kernel(const float* __restrict__ in, __bf16* __restrict__ out, int n4) {
  int i = blockIdx.x * blockDim.x + threadIdx.x;
  if (i < n4) {
    float4 f = ((const float4*)in)[i];
    B4U o;
    o.b[0] = (__bf16)f.x; o.b[1] = (__bf16)f.y;
    o.b[2] = (__bf16)f.z; o.b[3] = (__bf16)f.w;
    ((uint2*)out)[i] = o.u;
  }
}

// ---------------------------------------------------------------------------
// GEMM: C[M,N] = A[M,K]_bf16 * B[K,N]_bf16 + bias[N]_f32
// Workgroup tile 128x64, 8 waves (4x2), each wave 32x32 (2x2 WMMA tiles).
// K-step 64 => 8 WMMAs per barrier pair; A tile staged via async-to-LDS.
// ---------------------------------------------------------------------------
template <bool BF16OUT>
__global__ __launch_bounds__(256)
void gemm_kernel(const __bf16* __restrict__ A, const __bf16* __restrict__ B,
                 const float* __restrict__ bias, void* __restrict__ Cout,
                 int M, int N, int K) {
  __shared__ __bf16 sA[128 * 64];   // A tile, row-major [128][64]  (16 KB)
  __shared__ __bf16 sBt[64 * 64];   // B tile transposed [n][k]     (8 KB)

  const int t = threadIdx.x;
  const int lane = t & 31, wave = t >> 5;
  const int wm = wave & 3, wn = wave >> 2;          // 4x2 wave grid
  const int m0 = blockIdx.x * 128, n0 = blockIdx.y * 64;

  f32x8 acc[2][2] = {};

  for (int k0 = 0; k0 < K; k0 += 64) {
    __syncthreads();
    // A tile: 128x64 bf16, async 16B transactions (4 per thread).
#pragma unroll
    for (int i = 0; i < 4; ++i) {
      int lin = t + i * 256;
      int row = lin >> 3, c8 = (lin & 7) * 8;
      copy16_g2l(&A[(size_t)(m0 + row) * K + k0 + c8], &sA[row * 64 + c8]);
    }
    // B tile, transposed into LDS (2 x [vec8 read + 8 ds stores] per thread).
#pragma unroll
    for (int i = 0; i < 2; ++i) {
      int lin = t + i * 256;
      int kk = lin >> 3, n8 = (lin & 7) * 8;
      V8U v;
      v.q = *(const uint4*)&B[(size_t)(k0 + kk) * N + n0 + n8];
#pragma unroll
      for (int j = 0; j < 8; ++j) sBt[(n8 + j) * 64 + kk] = v.b[j];
    }
    // Prefetch next K-tile while this one is consumed.
    if (k0 + 64 < K) {
      if (t < 128)
        __builtin_prefetch(&A[(size_t)(m0 + t) * K + k0 + 64], 0, 0);
      else if (t < 192)
        __builtin_prefetch(&B[(size_t)(k0 + 64 + (t - 128)) * N + n0], 0, 0);
    }
    g2l_wait0();
    __syncthreads();

#pragma unroll
    for (int ks = 0; ks < 2; ++ks) {
      bf16x16 a0 = ld_frag_a(&sA[(wm * 32 + 0) * 64 + ks * 32], 64, lane);
      bf16x16 a1 = ld_frag_a(&sA[(wm * 32 + 16) * 64 + ks * 32], 64, lane);
      bf16x16 b0 = ld_frag_bt(&sBt[(wn * 32 + 0) * 64 + ks * 32], 64, lane);
      bf16x16 b1 = ld_frag_bt(&sBt[(wn * 32 + 16) * 64 + ks * 32], 64, lane);
      acc[0][0] = wmma_bf16(a0, b0, acc[0][0]);
      acc[0][1] = wmma_bf16(a0, b1, acc[0][1]);
      acc[1][0] = wmma_bf16(a1, b0, acc[1][0]);
      acc[1][1] = wmma_bf16(a1, b1, acc[1][1]);
    }
  }

  // Epilogue: C layout -> element (m = r + half*8, n = lane&15) per tile.
  const int half = lane >> 4, nn = lane & 15;
#pragma unroll
  for (int mt = 0; mt < 2; ++mt)
#pragma unroll
    for (int nt = 0; nt < 2; ++nt) {
      int gn = n0 + wn * 32 + nt * 16 + nn;
      float bv = bias[gn];
#pragma unroll
      for (int r = 0; r < 8; ++r) {
        int gm = m0 + wm * 32 + mt * 16 + half * 8 + r;
        float v = acc[mt][nt][r] + bv;
        if (BF16OUT)
          ((__bf16*)Cout)[(size_t)gm * N + gn] = (__bf16)v;
        else
          ((float*)Cout)[(size_t)gm * N + gn] = v;
      }
    }
}

// ---------------------------------------------------------------------------
// Sliding-window attention for one (b, h, q-block).
// LDS: Q[256][64], K[512][64], Vt[64][512], per-wave P scratch [16][64].
// Each wave processes 16-row M-tiles (mt = wave, wave+8), streaming 8 chunks
// of 64 keys with online (flash) softmax.
// ---------------------------------------------------------------------------
__global__ __launch_bounds__(256)
void attn_kernel(const __bf16* __restrict__ Qg, const __bf16* __restrict__ Kg,
                 const __bf16* __restrict__ Vg, __bf16* __restrict__ Og) {
  extern __shared__ __bf16 sm[];
  __bf16* sQ  = sm;                    // 256*64
  __bf16* sK  = sQ + 256 * 64;         // 512*64  (rows = key index)
  __bf16* sVt = sK + 512 * 64;         // 64*512  (rows = d, cols = key index)
  __bf16* sP  = sVt + 64 * 512;        // 8 waves * 16*64

  const int qb = blockIdx.x, h = blockIdx.y, b = blockIdx.z;
  const int t = threadIdx.x, lane = t & 31, wave = t >> 5;
  const size_t E = 1024, S = 4096;
  const size_t rowQ0 = (size_t)b * S + (size_t)qb * 256;
  const size_t rowK0 = rowQ0 - 256;    // only dereferenced when qb > 0
  const size_t colh  = (size_t)h * 64;

  // ---- Stage Q (async global->LDS) ----
#pragma unroll
  for (int i = 0; i < 8; ++i) {
    int lin = t + i * 256;
    int row = lin >> 3, c8 = (lin & 7) * 8;
    copy16_g2l(&Qg[(rowQ0 + row) * E + colh + c8], &sQ[row * 64 + c8]);
  }
  // ---- Stage K (prev + current block); zero-fill missing prev ----
#pragma unroll
  for (int i = 0; i < 16; ++i) {
    int lin = t + i * 256;
    int row = lin >> 3, c8 = (lin & 7) * 8;
    if (qb == 0 && row < 256) {
      uint4 z = {0u, 0u, 0u, 0u};
      *(uint4*)&sK[row * 64 + c8] = z;
    } else {
      copy16_g2l(&Kg[(rowK0 + row) * E + colh + c8], &sK[row * 64 + c8]);
    }
  }
  // ---- Stage V transposed (manual: permuting store) ----
#pragma unroll
  for (int i = 0; i < 16; ++i) {
    int lin = t + i * 256;
    int kv = lin >> 3, c8 = (lin & 7) * 8;
    V8U v;
    v.q = make_uint4(0u, 0u, 0u, 0u);
    if (!(qb == 0 && kv < 256))
      v.q = *(const uint4*)&Vg[(rowK0 + kv) * E + colh + c8];
#pragma unroll
    for (int j = 0; j < 8; ++j) sVt[(c8 + j) * 512 + kv] = v.b[j];
  }
  g2l_wait0();
  __syncthreads();

  __bf16* sPw = sP + wave * 16 * 64;
  const int half = lane >> 4, nn = lane & 15;
  const int minlo = (qb == 0) ? 256 : 0;   // uniform per block

  for (int mt = wave; mt < 16; mt += 8) {
    bf16x16 qa0 = ld_frag_a(&sQ[(mt * 16) * 64 + 0], 64, lane);
    bf16x16 qa1 = ld_frag_a(&sQ[(mt * 16) * 64 + 32], 64, lane);

    f32x8 O[4] = {};
    float rm[8], rs[8];
#pragma unroll
    for (int r = 0; r < 8; ++r) { rm[r] = -3.0e38f; rs[r] = 0.0f; }

    for (int ck = 0; ck < 8; ++ck) {
      // ---- S = Q * K^T for 64-key chunk (4 N-tiles x 2 K-steps) ----
      f32x8 Sc[4] = {};
#pragma unroll
      for (int nt = 0; nt < 4; ++nt) {
        const __bf16* kb = &sK[(ck * 64 + nt * 16) * 64];
        Sc[nt] = wmma_bf16(qa0, ld_frag_bt(kb + 0, 64, lane), Sc[nt]);
        Sc[nt] = wmma_bf16(qa1, ld_frag_bt(kb + 32, 64, lane), Sc[nt]);
      }

      // ---- mask + running row max (16-lane butterfly within half-wave) ----
      float mnew[8];
#pragma unroll
      for (int r = 0; r < 8; ++r) {
        int iq = mt * 16 + half * 8 + r;            // local query index
        int lo = iq + 1 > minlo ? iq + 1 : minlo;   // window start
        unsigned span = (unsigned)(iq + 256 - lo);  // window length - 1
        float mx = -3.0e38f;
#pragma unroll
        for (int nt = 0; nt < 4; ++nt) {
          int j = ck * 64 + nt * 16 + nn;           // local key index [0,512)
          bool ok = (unsigned)(j - lo) <= span;     // single range compare
          float s = ok ? Sc[nt][r] * 0.125f : -3.0e38f;  // 1/sqrt(64)
          Sc[nt][r] = s;
          mx = fmaxf(mx, s);
        }
        mx = fmaxf(mx, __shfl_xor(mx, 1));
        mx = fmaxf(mx, __shfl_xor(mx, 2));
        mx = fmaxf(mx, __shfl_xor(mx, 4));
        mx = fmaxf(mx, __shfl_xor(mx, 8));
        mnew[r] = fmaxf(rm[r], mx);
      }

      // ---- P = exp(S - m), update l, rescale O; stage P to LDS scratch ----
#pragma unroll
      for (int r = 0; r < 8; ++r) {
        float corr = __expf(rm[r] - mnew[r]);
        float psum = 0.0f;
#pragma unroll
        for (int nt = 0; nt < 4; ++nt) {
          float s = Sc[nt][r];
          float p = (s > -1.0e38f) ? __expf(s - mnew[r]) : 0.0f;
          psum += p;
          sPw[(half * 8 + r) * 64 + nt * 16 + nn] = (__bf16)p;
        }
        psum += __shfl_xor(psum, 1);
        psum += __shfl_xor(psum, 2);
        psum += __shfl_xor(psum, 4);
        psum += __shfl_xor(psum, 8);
        rs[r] = rs[r] * corr + psum;
        rm[r] = mnew[r];
#pragma unroll
        for (int dt = 0; dt < 4; ++dt) O[dt][r] *= corr;
      }

      // ---- O += P * V (P re-read from LDS in A-fragment layout) ----
      bf16x16 pa0 = ld_frag_a(sPw + 0, 64, lane);
      bf16x16 pa1 = ld_frag_a(sPw + 32, 64, lane);
#pragma unroll
      for (int dt = 0; dt < 4; ++dt) {
        const __bf16* vb = &sVt[(dt * 16) * 512 + ck * 64];
        O[dt] = wmma_bf16(pa0, ld_frag_bt(vb + 0, 512, lane), O[dt]);
        O[dt] = wmma_bf16(pa1, ld_frag_bt(vb + 32, 512, lane), O[dt]);
      }
    }

    // ---- normalize and store (bf16, [B,S,H*D] layout) ----
#pragma unroll
    for (int r = 0; r < 8; ++r) {
      float inv = 1.0f / rs[r];
      size_t g = (rowQ0 + mt * 16 + half * 8 + r) * E + colh;
#pragma unroll
      for (int dt = 0; dt < 4; ++dt)
        Og[g + dt * 16 + nn] = (__bf16)(O[dt][r] * inv);
    }
  }
}

// ---------------------------------------------------------------------------
// Host launcher
// ---------------------------------------------------------------------------
extern "C" void kernel_launch(void* const* d_in, const int* in_sizes, int n_in,
                              void* d_out, int out_size, void* d_ws, size_t ws_size,
                              hipStream_t stream) {
  (void)in_sizes; (void)n_in; (void)out_size; (void)ws_size;

  const float* q  = (const float*)d_in[0];
  const float* k  = (const float*)d_in[1];
  const float* v  = (const float*)d_in[2];
  const float* Wq = (const float*)d_in[3];
  const float* bq = (const float*)d_in[4];
  const float* Wk = (const float*)d_in[5];
  const float* bk = (const float*)d_in[6];
  const float* Wv = (const float*)d_in[7];
  const float* bv = (const float*)d_in[8];
  const float* Wo = (const float*)d_in[9];
  const float* bo = (const float*)d_in[10];

  const size_t NE = 2ull * 4096 * 1024;  // B*S*E = 8388608
  const size_t WE = 1024ull * 1024;      // E*E   = 1048576
  const int M = 8192, N = 1024, K = 1024;

  __bf16* p   = (__bf16*)d_ws;
  __bf16* Xq  = p; p += NE;
  __bf16* Xk  = p; p += NE;
  __bf16* Xv  = p; p += NE;
  __bf16* Wqb = p; p += WE;
  __bf16* Wkb = p; p += WE;
  __bf16* Wvb = p; p += WE;
  __bf16* Wob = p; p += WE;
  __bf16* Qb  = p; p += NE;
  __bf16* Kb  = p; p += NE;
  __bf16* Vb  = p; p += NE;
  __bf16* Ab  = p; p += NE;

  // f32 -> bf16 conversions
  cvt_kernel<<<(int)(NE / 4 / 256), 256, 0, stream>>>(q, Xq, (int)(NE / 4));
  cvt_kernel<<<(int)(NE / 4 / 256), 256, 0, stream>>>(k, Xk, (int)(NE / 4));
  cvt_kernel<<<(int)(NE / 4 / 256), 256, 0, stream>>>(v, Xv, (int)(NE / 4));
  cvt_kernel<<<(int)(WE / 4 / 256), 256, 0, stream>>>(Wq, Wqb, (int)(WE / 4));
  cvt_kernel<<<(int)(WE / 4 / 256), 256, 0, stream>>>(Wk, Wkb, (int)(WE / 4));
  cvt_kernel<<<(int)(WE / 4 / 256), 256, 0, stream>>>(Wv, Wvb, (int)(WE / 4));
  cvt_kernel<<<(int)(WE / 4 / 256), 256, 0, stream>>>(Wo, Wob, (int)(WE / 4));

  // Q/K/V projections (bf16 outputs, [B*S, E] head-major layout)
  dim3 gg(M / 128, N / 64);
  gemm_kernel<true><<<gg, 256, 0, stream>>>(Xq, Wqb, bq, (void*)Qb, M, N, K);
  gemm_kernel<true><<<gg, 256, 0, stream>>>(Xk, Wkb, bk, (void*)Kb, M, N, K);
  gemm_kernel<true><<<gg, 256, 0, stream>>>(Xv, Wvb, bv, (void*)Vb, M, N, K);

  // Sliding-window attention: one workgroup per (q-block, head, batch)
  const int smem = (256 * 64 + 512 * 64 + 64 * 512 + 8 * 16 * 64) * 2;  // 180224 B
  (void)hipFuncSetAttribute((const void*)attn_kernel,
                            hipFuncAttributeMaxDynamicSharedMemorySize, smem);
  attn_kernel<<<dim3(16, 16, 2), 256, smem, stream>>>(Qb, Kb, Vb, Ab);

  // Output projection (f32 output straight into d_out)
  gemm_kernel<false><<<gg, 256, 0, stream>>>(Ab, Wob, bo, d_out, M, N, K);
}
